// _LSTM_17377437679632
// MI455X (gfx1250) — matrix operand order repaired and checked by reading, not compiled, not run on hardware
//
#include <hip/hip_runtime.h>

typedef _Float16 v16h __attribute__((ext_vector_type(16)));
typedef _Float16 v8h  __attribute__((ext_vector_type(8)));
typedef float    v8f  __attribute__((ext_vector_type(8)));

#define B_ 512
#define S_ 512
#define F_ 64
#define H_ 128
#define O_ 64

// ---------------- small prep kernels ----------------

__global__ void k_f32_to_f16(const float* __restrict__ src, _Float16* __restrict__ dst, int n) {
  int i = blockIdx.x * blockDim.x + threadIdx.x;
  if (i < n) dst[i] = (_Float16)src[i];
}

__global__ void k_bias_sum(const float* __restrict__ a, const float* __restrict__ b,
                           float* __restrict__ o, int n) {
  int i = blockIdx.x * blockDim.x + threadIdx.x;
  if (i < n) o[i] = a[i] + b[i];
}

// ---------------- helpers ----------------

__device__ __forceinline__ v8f wmma_f16(v16h a, v16h b, v8f c) {
  // (neg_a, A, neg_b, B, c_mod, C, reuse_a, reuse_b)
  return __builtin_amdgcn_wmma_f32_16x16x32_f16(false, a, false, b, (short)0, c, false, false);
}

__device__ __forceinline__ v8f splat8f(float v) {
  v8f r;
#pragma unroll
  for (int e = 0; e < 8; ++e) r[e] = v;
  return r;
}

__device__ __forceinline__ v8h zero8h() {
  v8h r;
#pragma unroll
  for (int e = 0; e < 8; ++e) r[e] = (_Float16)0.0f;
  return r;
}

// Direct global -> LDS async copy, 16 bytes per lane (ASYNCcnt-tracked).
// VDST = VGPR with LDS byte address; VADDR = 64-bit global address.
__device__ __forceinline__ void async_load_b128(const _Float16* gsrc, _Float16* ldst) {
  unsigned long long ga = (unsigned long long)(uintptr_t)gsrc;
  unsigned int la = (unsigned int)(uintptr_t)ldst;   // low 32 bits = LDS offset
  asm volatile("global_load_async_to_lds_b128 %0, %1, off"
               :: "v"(la), "v"(ga)
               : "memory");
}

__device__ __forceinline__ void wait_asynccnt0() {
  asm volatile("s_wait_asynccnt 0x0" ::: "memory");
}

// A-operand (16x32 f16) per lane: m = lane&15, hi = lane>>4
// elems[0..15] = A[m][k0 + hi*16 + 0 .. +15]
__device__ __forceinline__ v16h load_a_tile(const _Float16* rowbase, int k0, int hi) {
  v16h r;
  r = *(const v16h*)(rowbase + k0 + hi * 16);
  return r;
}

// Branch-free activations: mul -> v_exp_f32 -> add -> v_rcp_f32.
// Raw hardware reciprocal instead of IEEE division (no div_scale/div_fixup
// sequence on the recurrence critical path).
__device__ __forceinline__ float sigm(float x) {
  return __builtin_amdgcn_rcpf(1.0f + __expf(-x));
}
__device__ __forceinline__ float tanh_fast(float x) {
  // tanh(x) = 2*sigmoid(2x) - 1 ; saturates correctly at +/-inf
  return __builtin_fmaf(2.0f, sigm(2.0f * x), -1.0f);
}

// ---------------- fused LSTM layer ----------------
// One block = 16 batch rows, 8 waves. Wave w owns gate cols j in [16w, 16w+16).
// Loops over all S timesteps; h/c stay on-chip; weights stay in VGPRs.
// x tiles are double-buffered in LDS via GLOBAL_LOAD_ASYNC_TO_LDS_B128.

template <int KIN, bool WRITE_SEQ, bool DO_FC>
__global__ __launch_bounds__(256)
void k_lstm(const _Float16* __restrict__ inSeq,  // [B][S][KIN] f16
            const _Float16* __restrict__ Wih,    // [4H][KIN]  f16 row-major
            const _Float16* __restrict__ Whh,    // [4H][H]    f16 row-major
            const float*    __restrict__ bias,   // [4H]  (b_ih + b_hh)
            _Float16* __restrict__ outSeq,       // [B][S][H] f16 (if WRITE_SEQ)
            const _Float16* __restrict__ fcw,    // [O][H] f16   (if DO_FC)
            const float*    __restrict__ fcb,    // [O]          (if DO_FC)
            float* __restrict__ out)             // [B][O]       (if DO_FC)
{
  constexpr int KX = KIN / 32;                 // A k-subtiles for input proj
  __shared__ __align__(32) _Float16 smX[2][16 * KIN];
  __shared__ __align__(32) _Float16 smH[16 * H_];

  const int tid   = threadIdx.x;
  const int w     = tid >> 5;
  const int lane  = tid & 31;
  const int laneN = lane & 15;
  const int hi    = lane >> 4;
  const int j0    = w * 16;
  const int b0    = blockIdx.x * 16;

  // ---- weight B-tiles resident in VGPRs ----
  // B-operand (32x16 f16) per lane: n = laneN, elems = W[n][k0 + hi*16 .. +15]
  v16h Bih[4][KX];
  v16h Bhh[4][4];
  float bv[4];
#pragma unroll
  for (int g = 0; g < 4; ++g) {
    const int n = g * H_ + j0 + laneN;
    bv[g] = bias[n];
#pragma unroll
    for (int kk = 0; kk < KX; ++kk)
      Bih[g][kk] = *(const v16h*)(Wih + (size_t)n * KIN + kk * 32 + hi * 16);
#pragma unroll
    for (int kk = 0; kk < 4; ++kk)
      Bhh[g][kk] = *(const v16h*)(Whh + (size_t)n * H_ + kk * 32 + hi * 16);
  }

  // ---- zero h in LDS ----
  for (int i = tid * 8; i < 16 * H_; i += 256 * 8)
    *(v8h*)(smH + i) = zero8h();

  // ---- staging map: 16B chunk per thread, one async instruction per active wave ----
  constexpr int CH = KIN / 8;                  // 8-half chunks per row
  const int srow = tid / CH;
  const int scol = (tid % CH) * 8;
  const bool sact = tid < 16 * CH;             // wave-uniform

  // stage tile 0 into buffer 0 (async, direct to LDS)
  if (sact)
    async_load_b128(inSeq + ((size_t)(b0 + srow) * S_ + 0) * KIN + scol,
                    &smX[0][srow * KIN + scol]);

  v8f cst = splat8f(0.0f);
  int cur = 0;

  for (int t = 0; t < S_; ++t) {
    wait_asynccnt0();   // issuing waves: tile t landed in smX[cur]
    __syncthreads();    // publish smX[cur] and h_{t-1} to all waves

    // async-prefetch x_{t+1} straight into the other LDS buffer
    const int tn = (t + 1 < S_) ? t + 1 : t;
    if (sact)
      async_load_b128(inSeq + ((size_t)(b0 + srow) * S_ + tn) * KIN + scol,
                      &smX[cur ^ 1][srow * KIN + scol]);

    // gates = bias + x_t @ Wih^T + h_{t-1} @ Whh^T   (each wave: 16 rows x 16 cols x 4 gates)
    v8f acc[4];
#pragma unroll
    for (int g = 0; g < 4; ++g) acc[g] = splat8f(bv[g]);

#pragma unroll
    for (int kk = 0; kk < KX; ++kk) {
      v16h ax = load_a_tile(&smX[cur][laneN * KIN], kk * 32, hi);
#pragma unroll
      for (int g = 0; g < 4; ++g) acc[g] = wmma_f16(ax, Bih[g][kk], acc[g]);
    }
#pragma unroll
    for (int kk = 0; kk < 4; ++kk) {
      v16h ah = load_a_tile(&smH[laneN * H_], kk * 32, hi);
#pragma unroll
      for (int g = 0; g < 4; ++g) acc[g] = wmma_f16(ah, Bhh[g][kk], acc[g]);
    }

    __syncthreads();  // all smH reads of h_{t-1} done

    // elementwise LSTM cell; D-layout: elem e -> row m = e + 8*hi, col j = j0+laneN
#pragma unroll
    for (int e = 0; e < 8; ++e) {
      float iv = sigm(acc[0][e]);
      float fv = sigm(acc[1][e]);
      float gv = tanh_fast(acc[2][e]);
      float ov = sigm(acc[3][e]);
      float cv = __builtin_fmaf(fv, cst[e], iv * gv);
      cst[e] = cv;
      float hv = ov * tanh_fast(cv);
      const int m = e + 8 * hi;
      _Float16 hh16 = (DO_FC && t == S_ - 1) ? (_Float16)fmaxf(hv, 0.0f) : (_Float16)hv;
      smH[m * H_ + j0 + laneN] = hh16;  // h_t for next step (relu'd on last step if feeding FC)
      if (WRITE_SEQ)
        outSeq[((size_t)(b0 + m) * S_ + t) * H_ + (j0 + laneN)] = (_Float16)hv;
    }
    cur ^= 1;
  }

  if (DO_FC) {
    __syncthreads();  // relu(h_last) f16 now in smH
    if (w < 4) {      // 4 waves cover O=64 output cols (wave-uniform branch, EXEC stays full)
      const int n0f = w * 16;
      v8f acc = splat8f(fcb[n0f + laneN]);
#pragma unroll
      for (int kk = 0; kk < 4; ++kk) {
        v16h ah = load_a_tile(&smH[laneN * H_], kk * 32, hi);
        v16h bf = *(const v16h*)(fcw + (size_t)(n0f + laneN) * H_ + kk * 32 + hi * 16);
        acc = wmma_f16(ah, bf, acc);
      }
#pragma unroll
      for (int e = 0; e < 8; ++e) {
        const int m = e + 8 * hi;
        out[(size_t)(b0 + m) * O_ + n0f + laneN] = acc[e];
      }
    }
  }
}

// ---------------- host launcher ----------------

extern "C" void kernel_launch(void* const* d_in, const int* in_sizes, int n_in,
                              void* d_out, int out_size, void* d_ws, size_t ws_size,
                              hipStream_t stream) {
  (void)in_sizes; (void)n_in; (void)out_size; (void)ws_size;

  const float* x    = (const float*)d_in[0];
  const float* Wih0 = (const float*)d_in[1];
  const float* Whh0 = (const float*)d_in[2];
  const float* bih0 = (const float*)d_in[3];
  const float* bhh0 = (const float*)d_in[4];
  const float* Wih1 = (const float*)d_in[5];
  const float* Whh1 = (const float*)d_in[6];
  const float* bih1 = (const float*)d_in[7];
  const float* bhh1 = (const float*)d_in[8];
  const float* fcw  = (const float*)d_in[9];
  const float* fcb  = (const float*)d_in[10];
  float* out = (float*)d_out;

  char* ws = (char*)d_ws;
  size_t off = 0;
  auto alloc = [&](size_t bytes) -> void* {
    void* p = ws + off;
    off = (off + bytes + 255) & ~(size_t)255;
    return p;
  };

  _Float16* xh    = (_Float16*)alloc((size_t)B_ * S_ * F_ * 2);  // 32 MB
  _Float16* h0h   = (_Float16*)alloc((size_t)B_ * S_ * H_ * 2);  // 64 MB
  _Float16* wih0h = (_Float16*)alloc((size_t)4 * H_ * F_ * 2);
  _Float16* whh0h = (_Float16*)alloc((size_t)4 * H_ * H_ * 2);
  _Float16* wih1h = (_Float16*)alloc((size_t)4 * H_ * H_ * 2);
  _Float16* whh1h = (_Float16*)alloc((size_t)4 * H_ * H_ * 2);
  _Float16* fcwh  = (_Float16*)alloc((size_t)O_ * H_ * 2);
  float*    bias0 = (float*)alloc((size_t)4 * H_ * 4);
  float*    bias1 = (float*)alloc((size_t)4 * H_ * 4);

  auto conv = [&](const float* s, _Float16* d, int n) {
    k_f32_to_f16<<<dim3((n + 255) / 256), dim3(256), 0, stream>>>(s, d, n);
  };
  conv(x,    xh,    B_ * S_ * F_);
  conv(Wih0, wih0h, 4 * H_ * F_);
  conv(Whh0, whh0h, 4 * H_ * H_);
  conv(Wih1, wih1h, 4 * H_ * H_);
  conv(Whh1, whh1h, 4 * H_ * H_);
  conv(fcw,  fcwh,  O_ * H_);
  k_bias_sum<<<dim3(2), dim3(256), 0, stream>>>(bih0, bhh0, bias0, 4 * H_);
  k_bias_sum<<<dim3(2), dim3(256), 0, stream>>>(bih1, bhh1, bias1, 4 * H_);

  // Layer 0: x -> h0_seq (f16)
  k_lstm<F_, true, false><<<dim3(B_ / 16), dim3(256), 0, stream>>>(
      xh, wih0h, whh0h, bias0, h0h, nullptr, nullptr, nullptr);

  // Layer 1 + ReLU + FC: h0_seq -> out (f32)
  k_lstm<H_, false, true><<<dim3(B_ / 16), dim3(256), 0, stream>>>(
      h0h, wih1h, whh1h, bias1, nullptr, fcwh, fcb, out);
}
